// Model_55783035240780
// MI455X (gfx1250) — matrix operand order repaired and checked by reading, not compiled, not run on hardware
//
#include <hip/hip_runtime.h>

// ---------------------------------------------------------------------------
// CDNA5 (gfx1250, wave32) GRU forecaster.
// Dominant cost: h @ Wh.T ([rows x 128] x [128 x 384]) per recurrent timestep
// -> v_wmma_f32_16x16x32_bf16, Wh^T fragments resident in VGPRs.
// This round: fast sigmoid/tanh (v_rcp_f32 / v_tanh_f32) on the latency-
// critical gate tail, and 2 barriers per timestep via double-buffered x.
// ---------------------------------------------------------------------------

typedef __attribute__((ext_vector_type(16))) __bf16 v16bf;
typedef __attribute__((ext_vector_type(8)))  __bf16 v8bf;
typedef __attribute__((ext_vector_type(8)))  float  v8f;

#define XSTRIDE 2052   // 2048 + PRED_LEN
#define DM      128
#define G3      384
#define NROWS   16     // rows per workgroup (one WMMA M-tile)
#define NT_PER_WAVE 3  // 24 N-tiles / 8 waves

// -------------------------------- helpers ----------------------------------

__device__ __forceinline__ unsigned short f32_to_bf16_bits(float f) {
    unsigned u = __float_as_uint(f);
    unsigned r = u + 0x7FFFu + ((u >> 16) & 1u);   // round-nearest-even
    return (unsigned short)(r >> 16);
}

__device__ __forceinline__ unsigned rotl32(unsigned v, int n) {
    return (v << n) | (v >> (32 - n));
}

// threefry2x32-style deterministic hash (stand-in for the JAX PRNG).
__device__ __forceinline__ void threefry2x32(unsigned k0, unsigned k1,
                                             unsigned& x0, unsigned& x1) {
    const unsigned ks[3] = { k0, k1, k0 ^ k1 ^ 0x1BD11BDAu };
    const int R[8] = { 13, 15, 26, 6, 17, 29, 16, 24 };
    x0 += ks[0]; x1 += ks[1];
    for (int i = 0; i < 20; ++i) {
        x0 += x1; x1 = rotl32(x1, R[i & 7]); x1 ^= x0;
        if ((i & 3) == 3) {
            int s = i / 4 + 1;
            x0 += ks[s % 3];
            x1 += ks[(s + 1) % 3] + (unsigned)s;
        }
    }
}

// fast 1/x (v_rcp_f32, no IEEE div fix-up chain)
__device__ __forceinline__ float fast_rcp(float x) {
    return __builtin_amdgcn_rcpf(x);
}

// sigmoid(x) = rcp(1 + exp(-x));  __expf lowers to v_exp_f32
__device__ __forceinline__ float fast_sigmoid(float x) {
    return fast_rcp(1.f + __expf(-x));
}

// tanh: hardware v_tanh_f32 on gfx1250 when available
__device__ __forceinline__ float fast_tanh(float x) {
#if __has_builtin(__builtin_amdgcn_tanhf)
    return __builtin_amdgcn_tanhf(x);
#else
    // tanh(x) = 1 - 2/(exp(2x)+1)
    return 1.f - 2.f * fast_rcp(__expf(2.f * x) + 1.f);
#endif
}

// -------------------------------- prep -------------------------------------
__global__ __launch_bounds__(256) void prep_kernel(
    const float* __restrict__ batch_x,
    const float* __restrict__ Wh_g, const float* __restrict__ Wh_w,
    float* __restrict__ x_ext,
    unsigned short* __restrict__ whT_g, unsigned short* __restrict__ whT_w) {
    int idx = blockIdx.x * 256 + threadIdx.x;
    const int NW = DM * G3;                       // 49152
    if (idx < NW) {
        int k = idx / G3, n = idx % G3;           // whT[k][n] = Wh[n][k]
        whT_g[idx] = f32_to_bf16_bits(Wh_g[n * DM + k]);
        whT_w[idx] = f32_to_bf16_bits(Wh_w[n * DM + k]);
    }
    if (idx < 64 * 2048) {
        int b = idx >> 11, t = idx & 2047;
        x_ext[b * XSTRIDE + t] = batch_x[idx];
    }
}

// ------------------------------ window starts ------------------------------
__global__ __launch_bounds__(256) void starts_kernel(int* __restrict__ starts,
                                                     int step, int T) {
    int idx = blockIdx.x * 256 + threadIdx.x;     // 0..4095 = b*64+m
    if (idx >= 64 * 64) return;
    unsigned x0 = 0x6A09E667u ^ (unsigned)step;
    unsigned x1 = (unsigned)idx;
    threefry2x32(0x243F6A88u, 0x85A308D3u ^ (unsigned)step, x0, x1);
    starts[idx] = (int)((x0 ^ x1) % (unsigned)(T - 64));
}

// ------------------------------ GRU (WMMA) ---------------------------------
// One workgroup recurs 16 rows.  256 threads = 8 wave32s; wave w owns gate
// column tiles [w*3, w*3+2].  Wh^T fragments live in VGPRs for all timesteps.
__global__ __launch_bounds__(256) void gru_wmma_kernel(
    const float* __restrict__ xext,
    const int*   __restrict__ startsPtr,          // nullptr => "main" mode
    const float* __restrict__ h0,                 // nullptr => zeros
    const unsigned short* __restrict__ whT,       // bf16 bits, [128][384]
    const float* __restrict__ Wi,                 // [384] (ENC_IN==1)
    const float* __restrict__ bi,
    const float* __restrict__ bh,
    float* __restrict__ Hout,                     // [rows][128]
    int seqLen) {
    __shared__ __align__(16) __bf16 sh_hbf[NROWS * DM];
    __shared__ float sh_hf[NROWS * DM];
    __shared__ float sh_gates[NROWS * G3];
    __shared__ float sh_wi[G3], sh_bi[G3], sh_bh[G3];
    __shared__ float sh_x[2][NROWS];              // double-buffered input col

    const int tid     = threadIdx.x;
    const int wave    = tid >> 5;
    const int lane    = tid & 31;
    const int rowBase = blockIdx.x * NROWS;

    // ---- resident B fragments: bf16 WhT, 32x16 per (K-chunk, N-tile) ------
    const __bf16* wT = reinterpret_cast<const __bf16*>(whT);
    const int bcol = lane & 15, bkh = lane >> 4;
    v16bf bfrag[NT_PER_WAVE][4];
#pragma unroll
    for (int j = 0; j < NT_PER_WAVE; ++j) {
        const int n = (wave * NT_PER_WAVE + j) * 16 + bcol;
#pragma unroll
        for (int kc = 0; kc < 4; ++kc) {
            const int kb = kc * 32 + bkh * 16;
            v16bf f;
#pragma unroll
            for (int e = 0; e < 16; ++e) f[e] = wT[(kb + e) * G3 + n];
            bfrag[j][kc] = f;
        }
    }

    // ---- init hidden state + stage small weights in LDS -------------------
    for (int w = tid; w < G3; w += 256) {
        sh_wi[w] = Wi[w]; sh_bi[w] = bi[w]; sh_bh[w] = bh[w];
    }
#pragma unroll
    for (int p = 0; p < 8; ++p) {
        int idx = tid + p * 256;                  // 16*128 = 2048 elems
        int row = idx >> 7, d = idx & 127;
        float hv = h0 ? h0[(rowBase + row) * DM + d] : 0.f;
        sh_hf[idx]  = hv;
        sh_hbf[idx] = (__bf16)hv;
    }
    // preload x column for t = 0
    if (tid < NROWS) {
        int gr = rowBase + tid;
        int xi = startsPtr ? ((gr >> 6) * XSTRIDE + startsPtr[gr])
                           : (gr * XSTRIDE);
        sh_x[0][tid] = xext[xi];
    }
    __syncthreads();

    const int arow = lane & 15, ahalf = lane >> 4;   // A-matrix lane mapping
    const int ccol = lane & 15, chalf = lane >> 4;   // C-matrix lane mapping

    for (int t = 0; t < seqLen; ++t) {
        const int cur = t & 1, nxt = cur ^ 1;

        // A fragments from LDS bf16 hidden state (contiguous K order:
        // lanes<16 hold K{0..15}, lanes>=16 hold K{16..31}).
        v16bf afrag[4];
#pragma unroll
        for (int kc = 0; kc < 4; ++kc) {
            union { v16bf v; v8bf h[2]; } ua;
            ua.h[0] = *reinterpret_cast<const v8bf*>(
                &sh_hbf[arow * DM + kc * 32 + ahalf * 16]);
            ua.h[1] = *reinterpret_cast<const v8bf*>(
                &sh_hbf[arow * DM + kc * 32 + ahalf * 16 + 8]);
            afrag[kc] = ua.v;
        }

        // gates_pre = h @ Wh^T  via 3 N-tiles x 4 K-chunks of WMMA
#pragma unroll
        for (int j = 0; j < NT_PER_WAVE; ++j) {
            v8f acc = { 0.f, 0.f, 0.f, 0.f, 0.f, 0.f, 0.f, 0.f };
#pragma unroll
            for (int kc = 0; kc < 4; ++kc)
                acc = __builtin_amdgcn_wmma_f32_16x16x32_bf16(
                    false, afrag[kc], false, bfrag[j][kc],
                    (short)0, acc, false, false);
            const int ntile = wave * NT_PER_WAVE + j;
#pragma unroll
            for (int e = 0; e < 8; ++e)
                sh_gates[(e + 8 * chalf) * G3 + ntile * 16 + ccol] = acc[e];
        }
        __syncthreads();   // gates visible; sh_hbf free to be overwritten

        // overlap: fetch next step's input column while updating gates
        if (tid < NROWS && t + 1 < seqLen) {
            int gr = rowBase + tid;
            int xi = startsPtr
                   ? ((gr >> 6) * XSTRIDE + startsPtr[gr] + t + 1)
                   : (gr * XSTRIDE + t + 1);
            sh_x[nxt][tid] = xext[xi];
            if (!startsPtr && t + 16 < seqLen)
                __builtin_prefetch(&xext[gr * XSTRIDE + t + 16], 0, 0);
        }

        // GRU cell update (gate order r,z,n), 2048 elems / 256 threads
#pragma unroll
        for (int p = 0; p < 8; ++p) {
            int idx = tid + p * 256;
            int row = idx >> 7, d = idx & 127;
            float xv  = sh_x[cur][row];
            float gr_ = xv * sh_wi[d]       + sh_bi[d]
                      + sh_gates[row * G3 + d]       + sh_bh[d];
            float gz_ = xv * sh_wi[d + 128] + sh_bi[d + 128]
                      + sh_gates[row * G3 + d + 128] + sh_bh[d + 128];
            float gnh = sh_gates[row * G3 + d + 256] + sh_bh[d + 256];
            float gni = xv * sh_wi[d + 256] + sh_bi[d + 256];
            float r = fast_sigmoid(gr_);
            float z = fast_sigmoid(gz_);
            float n = fast_tanh(gni + r * gnh);
            float hN = (1.f - z) * n + z * sh_hf[idx];
            sh_hf[idx]  = hN;
            sh_hbf[idx] = (__bf16)hN;
        }
        __syncthreads();   // h ready for next A-build; x[nxt] ready
    }

#pragma unroll
    for (int p = 0; p < 8; ++p) {
        int idx = tid + p * 256;
        int row = idx >> 7, d = idx & 127;
        Hout[(rowBase + row) * DM + d] = sh_hf[idx];
    }
}

// --------------------------- attention / head ------------------------------
__global__ __launch_bounds__(256) void attn_kernel(
    float* __restrict__ x_ext,
    const float* __restrict__ H, const float* __restrict__ S,
    const int* __restrict__ starts,
    const float* __restrict__ Wd, const float* __restrict__ bd,
    const float* __restrict__ Wc, const float* __restrict__ bc,
    float* __restrict__ us, int T, int step) {
    const int b = blockIdx.x;
    __shared__ float red[256], red2[256], lg[64], stats[2];
    float* xr = x_ext + b * XSTRIDE;

    float s = 0.f, s2 = 0.f;
    for (int t = threadIdx.x; t < T; t += 256) { float v = xr[t]; s += v; s2 += v * v; }
    red[threadIdx.x] = s; red2[threadIdx.x] = s2;
    __syncthreads();
    for (int off = 128; off > 0; off >>= 1) {
        if (threadIdx.x < off) {
            red[threadIdx.x]  += red[threadIdx.x + off];
            red2[threadIdx.x] += red2[threadIdx.x + off];
        }
        __syncthreads();
    }
    if (threadIdx.x == 0) {
        float mean = red[0] / (float)T;
        float var  = (red2[0] - mean * red[0]) / (float)(T - 1);  // unbiased
        stats[0] = mean; stats[1] = sqrtf(fmaxf(var, 0.f));
    }
    if (threadIdx.x < 64) {                        // logits over 64 windows
        float acc = 0.f;
        for (int k = 0; k < DM; ++k)
            acc += H[b * DM + k] * S[(b * 64 + threadIdx.x) * DM + k];
        lg[threadIdx.x] = acc;
    }
    __syncthreads();
    if (threadIdx.x == 0) {
        float m = -1e30f;
        for (int i = 0; i < 64; ++i) m = fmaxf(m, lg[i]);
        float den = 0.f;
        for (int i = 0; i < 64; ++i) { lg[i] = __expf(lg[i] - m); den += lg[i]; }
        float inv_den = fast_rcp(den);
        float thr = stats[0] + 1.48f * stats[1];
        float sA = 0.f;
        for (int i = 0; i < 64; ++i) {
            float A = lg[i] * inv_den;
            float q = (xr[starts[b * 64 + i] + 64] > thr) ? 1.f : 0.f;
            sA += q * A;
        }
        float u = fast_sigmoid(sA * Wc[0] + bc[0]);
        float o = bd[0];
        for (int k = 0; k < DM; ++k) o += H[b * DM + k] * Wd[k];
        xr[T] = o + u;                              // append prediction
        us[b * 4 + step] = u;
    }
}

// -------------------------------- output -----------------------------------
__global__ __launch_bounds__(256) void finalize_kernel(
    const float* __restrict__ x_ext, const float* __restrict__ us,
    float* __restrict__ out) {
    int i = blockIdx.x * 256 + threadIdx.x;
    if (i < 256) {                                  // x[:, -4:, :]
        int b = i >> 2, k = i & 3;
        out[i] = x_ext[b * XSTRIDE + 2048 + k];
    } else if (i < 512) {
        out[i] = us[i - 256];                       // concatenated u's
    }
}

// -------------------------------- launch -----------------------------------
extern "C" void kernel_launch(void* const* d_in, const int* in_sizes, int n_in,
                              void* d_out, int out_size, void* d_ws, size_t ws_size,
                              hipStream_t stream) {
    const float* batch_x = (const float*)d_in[0];
    const float* Wi_g = (const float*)d_in[4];
    const float* Wh_g = (const float*)d_in[5];
    const float* bi_g = (const float*)d_in[6];
    const float* bh_g = (const float*)d_in[7];
    const float* Wi_w = (const float*)d_in[8];
    const float* Wh_w = (const float*)d_in[9];
    const float* bi_w = (const float*)d_in[10];
    const float* bh_w = (const float*)d_in[11];
    const float* Wd   = (const float*)d_in[12];
    const float* bd   = (const float*)d_in[13];
    const float* Wc   = (const float*)d_in[16];
    const float* bc   = (const float*)d_in[17];
    float* out = (float*)d_out;

    // workspace carve-up (~2.9 MB)
    float* x_ext = (float*)d_ws;                    // 64*2052
    float* H     = x_ext + 64 * XSTRIDE;            // 64*128
    float* S     = H + 64 * DM;                     // 4096*128
    float* us    = S + 4096 * DM;                   // 256
    int*   starts = (int*)(us + 256);               // 4096
    unsigned short* whT_g = (unsigned short*)(starts + 4096);  // 49152
    unsigned short* whT_w = whT_g + DM * G3;                   // 49152

    prep_kernel<<<512, 256, 0, stream>>>(batch_x, Wh_g, Wh_w, x_ext, whT_g, whT_w);

    for (int i = 0; i < 4; ++i) {
        const int T = 2048 + i;
        const int   seqMain = (i == 0) ? T : (T - 1);
        const float* h0     = (i == 0) ? nullptr : H;

        starts_kernel<<<16, 256, 0, stream>>>(starts, i, T);
        gru_wmma_kernel<<<64 / NROWS, 256, 0, stream>>>(
            x_ext, nullptr, h0, whT_g, Wi_g, bi_g, bh_g, H, seqMain);
        gru_wmma_kernel<<<4096 / NROWS, 256, 0, stream>>>(
            x_ext, starts, nullptr, whT_w, Wi_w, bi_w, bh_w, S, 64);
        attn_kernel<<<64, 256, 0, stream>>>(
            x_ext, H, S, starts, Wd, bd, Wc, bc, us, T, i);
    }

    finalize_kernel<<<2, 256, 0, stream>>>(x_ext, us, out);
}
